// LSTMClassifier_17102559772819
// MI455X (gfx1250) — compile-verified, matrix operand order checked by reading, not run on hardware
//
#include <hip/hip_runtime.h>
#include <math.h>

#define HN 512
#define EN 300
#define EPAD 320          // 300 padded to multiple of 32
#define KTOT 832          // 812 padded to multiple of 32
#define KREAL 812
#define BB 256
#define TT 200
#define PAD_IDX 1
#define HPAD 520          // LDS h-row stride in shorts (bank-conflict pad, 16B aligned)

#if __has_builtin(__builtin_amdgcn_sched_barrier)
#define SCHED_FENCE() __builtin_amdgcn_sched_barrier(0)
#else
#define SCHED_FENCE()
#endif

typedef __attribute__((ext_vector_type(16))) __bf16 v16bf;
typedef __attribute__((ext_vector_type(8)))  float  v8f;

static __device__ __forceinline__ unsigned short f2bf(float f) {
    unsigned int u = __float_as_uint(f);
    unsigned int r = u + 0x7FFFu + ((u >> 16) & 1u);   // round-to-nearest-even
    return (unsigned short)(r >> 16);
}

union FragBF {
    v16bf v;
    unsigned short s[16];
    uint4 q[2];
};

// Fallback: 8 embedding values (fp32 -> bf16) from one embed row, zero pad past 300.
static __device__ __forceinline__ void load8_embed(const float* __restrict__ row,
                                                   int s, unsigned short* dst) {
    if (s + 8 <= EN) {
        float4 a = *(const float4*)(row + s);
        float4 b = *(const float4*)(row + s + 4);
        dst[0] = f2bf(a.x); dst[1] = f2bf(a.y); dst[2] = f2bf(a.z); dst[3] = f2bf(a.w);
        dst[4] = f2bf(b.x); dst[5] = f2bf(b.y); dst[6] = f2bf(b.z); dst[7] = f2bf(b.w);
    } else {
#pragma unroll
        for (int j = 0; j < 8; ++j) {
            int e = s + j;
            dst[j] = (e < EN) ? f2bf(row[e]) : (unsigned short)0;
        }
    }
}

static __device__ __forceinline__ void load_b16(FragBF& f, const unsigned short* row,
                                                int ka0, int ka1) {
    f.q[0] = *(const uint4*)(row + ka0);
    f.q[1] = *(const uint4*)(row + ka1);
}

// ---------------- init: zero final buffer, pack biases, compute lengths ----------------
__global__ void init_kernel(float* fin, float* bc, int* len,
                            const float* bi, const float* bfv, const float* bo, const float* bg,
                            const int* x) {
    int id = blockIdx.x * blockDim.x + threadIdx.x;
    if (id < BB * HN) fin[id] = 0.f;
    if (id < 4 * HN) {
        int gi = id / HN, n = id % HN;
        const float* bp = (gi == 0) ? bi : (gi == 1) ? bfv : (gi == 2) ? bo : bg;
        bc[id] = bp[n];
    }
    if (id < BB) {
        int cnt = 0;
        for (int t = 0; t < TT; ++t) cnt += (x[id * TT + t] != PAD_IDX);
        len[id] = cnt;
    }
}

// ---------------- pack weights: fp32 [512][812] x4 -> bf16 Wc[2048][832] ----------------
__global__ void pack_w_kernel(unsigned short* Wc, const float* Wi, const float* Wf,
                              const float* Wo, const float* Wg) {
    int id = blockIdx.x * blockDim.x + threadIdx.x;
    if (id >= 4 * HN * KTOT) return;
    int gi  = id / (HN * KTOT);
    int rem = id % (HN * KTOT);
    int n = rem / KTOT, k = rem % KTOT;
    const float* W = (gi == 0) ? Wi : (gi == 1) ? Wf : (gi == 2) ? Wo : Wg;
    float v = (k < KREAL) ? W[(size_t)n * KREAL + k] : 0.f;
    Wc[id] = f2bf(v);
}

// ---------------- gather embeddings to bf16: xe[t][b][320] ----------------
__global__ void pack_xe_kernel(unsigned short* xe, const float* __restrict__ embed,
                               const int* __restrict__ x) {
    int id = blockIdx.x * blockDim.x + threadIdx.x;
    if (id >= TT * BB * EPAD) return;
    int e  = id % EPAD;
    int bt = id / EPAD;
    int b  = bt % BB;
    int t  = bt / BB;
    float v = 0.f;
    if (e < EN) {
        int row = x[b * TT + t];
        v = embed[(size_t)row * EN + e];
    }
    xe[id] = f2bf(v);
}

// -------- parallel input-projection GEMM: preact[t*B+b][2048] = xe @ WcE^T --------
__global__ void __launch_bounds__(128)
embed_gemm_kernel(const unsigned short* __restrict__ Wc, const unsigned short* __restrict__ xe,
                  float* __restrict__ preact) {
    const int lane = threadIdx.x;
    const int wid  = blockIdx.x * 4 + threadIdx.y;
    const int n_tile = wid & 31;
    const int mt     = wid >> 5;          // 0..3199 (51200 rows / 16)
    const int lr   = lane & 15;
    const int half = lane >> 4;
    const int n0 = n_tile * 16;

    const unsigned short* arow  = xe + (size_t)(mt * 16 + lr) * EPAD;
    const unsigned short* brow0 = Wc + (size_t)(0 * HN + n0 + lr) * KTOT + HN; // e-part cols
    const unsigned short* brow1 = Wc + (size_t)(1 * HN + n0 + lr) * KTOT + HN;
    const unsigned short* brow2 = Wc + (size_t)(2 * HN + n0 + lr) * KTOT + HN;
    const unsigned short* brow3 = Wc + (size_t)(3 * HN + n0 + lr) * KTOT + HN;

    v8f acc0 = {}, acc1 = {}, acc2 = {}, acc3 = {};
    FragBF a, b0, b1, b2, b3;
    load_b16(a, arow, half * 8, 16 + half * 8);
    load_b16(b0, brow0, half * 8, 16 + half * 8);
    load_b16(b1, brow1, half * 8, 16 + half * 8);
    load_b16(b2, brow2, half * 8, 16 + half * 8);
    load_b16(b3, brow3, half * 8, 16 + half * 8);

    for (int kk = 0; kk < EPAD / 32; ++kk) {
        const int kn  = (kk + 1 < EPAD / 32) ? kk + 1 : kk;
        const int ka0 = kn * 32 + half * 8;
        const int ka1 = kn * 32 + 16 + half * 8;
        FragBF an, b0n, b1n, b2n, b3n;
        load_b16(an, arow, ka0, ka1);
        load_b16(b0n, brow0, ka0, ka1);
        load_b16(b1n, brow1, ka0, ka1);
        load_b16(b2n, brow2, ka0, ka1);
        load_b16(b3n, brow3, ka0, ka1);
        SCHED_FENCE();
        acc0 = __builtin_amdgcn_wmma_f32_16x16x32_bf16(false, a.v, false, b0.v, (short)0, acc0, false, false);
        acc1 = __builtin_amdgcn_wmma_f32_16x16x32_bf16(false, a.v, false, b1.v, (short)0, acc1, false, false);
        acc2 = __builtin_amdgcn_wmma_f32_16x16x32_bf16(false, a.v, false, b2.v, (short)0, acc2, false, false);
        acc3 = __builtin_amdgcn_wmma_f32_16x16x32_bf16(false, a.v, false, b3.v, (short)0, acc3, false, false);
        a = an; b0 = b0n; b1 = b1n; b2 = b2n; b3 = b3n;
    }

#pragma unroll
    for (int r = 0; r < 8; ++r) {
        const int m = mt * 16 + half * 8 + r;
        const size_t base = (size_t)m * (4 * HN) + n0 + lr;
        preact[base + 0 * HN] = acc0[r];
        preact[base + 1 * HN] = acc1[r];
        preact[base + 2 * HN] = acc2[r];
        preact[base + 3 * HN] = acc3[r];
    }
}

// ---------------- persistent LSTM sequence kernel ----------------
// One workgroup (32 waves = 1024 threads) owns one M-tile (16 batch rows) for all
// 200 timesteps. h is double-buffered in LDS, c stays in VGPRs, weights stream
// from L2. Workgroup barrier between steps.
__global__ void __launch_bounds__(1024)
lstm_seq_kernel(const unsigned short* __restrict__ Wc, const float* __restrict__ bc,
                float* __restrict__ fin, const int* __restrict__ x,
                const int* __restrict__ len, const float* __restrict__ embed,
                const unsigned short* __restrict__ xe, const float* __restrict__ preact,
                int nk) {
    __shared__ unsigned short hsh[2][16 * HPAD];

    const int lane = threadIdx.x;
    const int w    = threadIdx.y;        // n_tile 0..31
    const int mt   = blockIdx.x;         // m_tile 0..15
    const int lr   = lane & 15;
    const int half = lane >> 4;
    const int m0 = mt * 16;
    const int n0 = w * 16;
    const int n  = n0 + lr;

    // zero h(t=0) buffer
    for (int i = w * 32 + lane; i < 16 * HPAD; i += 1024) hsh[0][i] = 0;

    const unsigned short* brow0 = Wc + (size_t)(0 * HN + n) * KTOT;
    const unsigned short* brow1 = Wc + (size_t)(1 * HN + n) * KTOT;
    const unsigned short* brow2 = Wc + (size_t)(2 * HN + n) * KTOT;
    const unsigned short* brow3 = Wc + (size_t)(3 * HN + n) * KTOT;

    const float bci = bc[0 * HN + n];
    const float bcf = bc[1 * HN + n];
    const float bco = bc[2 * HN + n];
    const float bcg = bc[3 * HN + n];

    int lenr[8];
#pragma unroll
    for (int r = 0; r < 8; ++r) lenr[r] = len[m0 + half * 8 + r];

    float creg[8];
#pragma unroll
    for (int r = 0; r < 8; ++r) creg[r] = 0.f;

    __syncthreads();

    for (int t = 0; t < TT; ++t) {
        const unsigned short* hbuf = &hsh[t & 1][0];
        unsigned short*      hnext = &hsh[(t & 1) ^ 1][0];
        const unsigned short* xerow =
            xe ? xe + (size_t)(t * BB + m0 + lr) * EPAD : (const unsigned short*)0;
        const float* emrow =
            xe ? (const float*)0 : embed + (size_t)x[(m0 + lr) * TT + t] * EN;

        // Accumulator init: precomputed e-projection (fp32) or zero.
        v8f acc0 = {}, acc1 = {}, acc2 = {}, acc3 = {};
        if (preact) {
            const float* prow = preact + (size_t)(t * BB) * (4 * HN);
#pragma unroll
            for (int r = 0; r < 8; ++r) {
                const size_t base = (size_t)(m0 + half * 8 + r) * (4 * HN) + n0 + lr;
                acc0[r] = prow[base + 0 * HN];
                acc1[r] = prow[base + 1 * HN];
                acc2[r] = prow[base + 2 * HN];
                acc3[r] = prow[base + 3 * HN];
            }
        }

        // A fragment: h-part from LDS, e-part from xe (global) or raw embed fallback.
        auto load_a = [&](FragBF& a, int kb) {
            const int ka0 = kb + half * 8;
            const int ka1 = kb + 16 + half * 8;
            if (kb < HN) {
                const unsigned short* hrow = hbuf + lr * HPAD;
                a.q[0] = *(const uint4*)(hrow + ka0);
                a.q[1] = *(const uint4*)(hrow + ka1);
            } else if (xerow) {
                load_b16(a, xerow, ka0 - HN, ka1 - HN);
            } else {
                load8_embed(emrow, ka0 - HN, &a.s[0]);
                load8_embed(emrow, ka1 - HN, &a.s[8]);
            }
        };

        FragBF a, b0, b1, b2, b3;
        load_a(a, 0);
        load_b16(b0, brow0, half * 8, 16 + half * 8);
        load_b16(b1, brow1, half * 8, 16 + half * 8);
        load_b16(b2, brow2, half * 8, 16 + half * 8);
        load_b16(b3, brow3, half * 8, 16 + half * 8);

        for (int kk = 0; kk < nk; ++kk) {
            const int kn  = (kk + 1 < nk) ? kk + 1 : kk;
            const int kb  = kn * 32;
            const int ka0 = kb + half * 8;
            const int ka1 = kb + 16 + half * 8;
            FragBF an, b0n, b1n, b2n, b3n;
            load_a(an, kb);
            load_b16(b0n, brow0, ka0, ka1);
            load_b16(b1n, brow1, ka0, ka1);
            load_b16(b2n, brow2, ka0, ka1);
            load_b16(b3n, brow3, ka0, ka1);
            SCHED_FENCE();
            acc0 = __builtin_amdgcn_wmma_f32_16x16x32_bf16(false, a.v, false, b0.v, (short)0, acc0, false, false);
            acc1 = __builtin_amdgcn_wmma_f32_16x16x32_bf16(false, a.v, false, b1.v, (short)0, acc1, false, false);
            acc2 = __builtin_amdgcn_wmma_f32_16x16x32_bf16(false, a.v, false, b2.v, (short)0, acc2, false, false);
            acc3 = __builtin_amdgcn_wmma_f32_16x16x32_bf16(false, a.v, false, b3.v, (short)0, acc3, false, false);
            a = an; b0 = b0n; b1 = b1n; b2 = b2n; b3 = b3n;
        }

        // Epilogue: gates + c/h update. Lane holds fixed column n, 8 rows.
#pragma unroll
        for (int r = 0; r < 8; ++r) {
            const int mloc = half * 8 + r;            // row within the tile
            float pi = acc0[r] + bci;
            float pf = acc1[r] + bcf;
            float po = acc2[r] + bco;
            float pg = acc3[r] + bcg;
            float ig = 1.f / (1.f + __expf(-pi));
            float fg = 1.f / (1.f + __expf(-pf));
            float og = 1.f / (1.f + __expf(-po));
            float gg = tanhf(pg);
            float cn = fg * creg[r] + ig * gg;
            creg[r] = cn;
            float h = tanhf(cn) * og;
            hnext[mloc * HPAD + n] = f2bf(h);
            if (t == lenr[r] - 1) {
                const int m = m0 + mloc;
                fin[m * HN + n] = (x[m * TT + t] == PAD_IDX) ? 0.f : h;
            }
        }
        __syncthreads();
    }
}

// ---------------- final projection: [256,512] x [512,5] + bias ----------------
__global__ void final_proj_kernel(const float* __restrict__ fin,
                                  const float* __restrict__ Wout,
                                  const float* __restrict__ bout,
                                  float* __restrict__ out) {
    int b = threadIdx.x;
    float acc[5] = {0.f, 0.f, 0.f, 0.f, 0.f};
    for (int h = 0; h < HN; ++h) {
        float fh = fin[b * HN + h];
#pragma unroll
        for (int o = 0; o < 5; ++o) acc[o] += fh * Wout[o * HN + h];
    }
#pragma unroll
    for (int o = 0; o < 5; ++o) out[b * 5 + o] = acc[o] + bout[o];
}

extern "C" void kernel_launch(void* const* d_in, const int* in_sizes, int n_in,
                              void* d_out, int out_size, void* d_ws, size_t ws_size,
                              hipStream_t stream) {
    (void)in_sizes; (void)n_in; (void)out_size;
    const int*   x     = (const int*)d_in[0];
    const float* embed = (const float*)d_in[1];
    const float* Wi    = (const float*)d_in[2];
    const float* bi    = (const float*)d_in[3];
    const float* Wf    = (const float*)d_in[4];
    const float* bfv   = (const float*)d_in[5];
    const float* Wg    = (const float*)d_in[6];
    const float* bg    = (const float*)d_in[7];
    const float* Wo    = (const float*)d_in[8];
    const float* bo    = (const float*)d_in[9];
    const float* Wout  = (const float*)d_in[10];
    const float* bout  = (const float*)d_in[11];

    char* ws = (char*)d_ws;
    auto al = [](size_t v) { return (v + 255) & ~(size_t)255; };
    size_t off = 0;
    unsigned short* Wc  = (unsigned short*)(ws + off); off = al(off + (size_t)4 * HN * KTOT * 2);
    float*          fin = (float*)(ws + off);          off = al(off + (size_t)BB * HN * 4);
    float*          bc  = (float*)(ws + off);          off = al(off + (size_t)4 * HN * 4);
    int*            len = (int*)(ws + off);            off = al(off + (size_t)BB * 4);
    unsigned short* xe  = (unsigned short*)(ws + off);
    size_t off_xe  = al(off + (size_t)TT * BB * EPAD * 2);
    float* preact = (float*)(ws + off_xe);
    size_t off_pre = al(off_xe + (size_t)TT * BB * 4 * HN * 4);

    const bool has_xe  = ws_size >= off_xe;
    const bool has_pre = ws_size >= off_pre;

    init_kernel<<<(BB * HN + 255) / 256, 256, 0, stream>>>(fin, bc, len, bi, bfv, bo, bg, x);
    pack_w_kernel<<<(4 * HN * KTOT + 255) / 256, 256, 0, stream>>>(Wc, Wi, Wf, Wo, Wg);
    if (has_xe) {
        pack_xe_kernel<<<(TT * BB * EPAD + 255) / 256, 256, 0, stream>>>(xe, embed, x);
    }
    if (has_pre) {
        embed_gemm_kernel<<<3200 * 32 / 4, dim3(32, 4), 0, stream>>>(Wc, xe, preact);
    }

    const int nk = has_pre ? (HN / 32) : (KTOT / 32);
    lstm_seq_kernel<<<16, dim3(32, 32), 0, stream>>>(
        Wc, bc, fin, x, len, embed,
        has_xe ? xe : (const unsigned short*)0,
        has_pre ? preact : (const float*)0, nk);

    final_proj_kernel<<<1, 256, 0, stream>>>(fin, Wout, bout, (float*)d_out);
}
